// PointManifold2_partseg_pointchd_32298154066462
// MI455X (gfx1250) — compile-verified, matrix operand order checked by reading, not run on hardware
//
#include <hip/hip_runtime.h>
#include <hip/hip_bf16.h>

typedef __attribute__((ext_vector_type(16))) __bf16 v16bf;
typedef __attribute__((ext_vector_type(8)))  float  v8f;

#define BATCH 8
#define NPTS  2048
#define KNN   20
#define NEGSL 0.2f
#define NINF  (-1e30f)

__device__ __forceinline__ float lrelu(float x){ return x >= 0.0f ? x : NEGSL*x; }

__device__ __forceinline__ v8f wmma_bf16(v16bf a, v16bf b, v8f c){
  return __builtin_amdgcn_wmma_f32_16x16x32_bf16(false, a, false, b, (short)0, c, false, false);
}

// Load one 16x32 bf16 operand tile (A: rows=M, B: rows=N) from a row-major
// matrix with leading dimension ld (elements). Per CDNA5 ISA layout:
// lane L: row = L%16, half h = L/16; elems 0-7 <- K = h*8+0..7, elems 8-15 <- K = 16+h*8+0..7.
__device__ __forceinline__ v16bf load_tile16(const __bf16* base, int ld, int lane){
  int r = lane & 15, h = (lane >> 4) & 1;
  const __bf16* p = base + r*ld + h*8;
  v16bf v;
#pragma unroll
  for (int i = 0; i < 8; ++i){ v[i] = p[i]; v[8+i] = p[16+i]; }
  return v;
}

// Native CDNA5 float-max atomics (no-return forms; single instruction, no CAS loop).
__device__ __forceinline__ void dsMaxF(float* p, float v){
  unsigned off = (unsigned)(size_t)p;            // flat LDS addr[31:0] == LDS byte offset
  asm volatile("ds_max_num_f32 %0, %1" :: "v"(off), "v"(v) : "memory");
}
__device__ __forceinline__ void globalMaxF(float* p, float v){
  asm volatile("global_atomic_max_num_f32 %0, %1, off" :: "v"(p), "v"(v) : "memory");
}
__device__ __forceinline__ void waitDS(){ asm volatile("s_wait_dscnt 0x0" ::: "memory"); }
__device__ __forceinline__ void waitAsync(){ asm volatile("s_wait_asynccnt 0x0" ::: "memory"); }

// CDNA5 async global->LDS copy, 16 bytes per lane, tracked by ASYNCcnt.
__device__ __forceinline__ void async_g2l_b128(unsigned lds_off, const void* gaddr){
  asm volatile("global_load_async_to_lds_b128 %0, %1, off" :: "v"(lds_off), "v"(gaddr) : "memory");
}

// ---------------------------------------------------------------- prep
__global__ __launch_bounds__(256) void prep_kernel(
    const float* w_mp1, const float* s_mp1, const float* b_mp1,
    const float* w_mp2, const float* s_mp2, const float* b_mp2,
    const float* w_mp3, const float* s_mp3, const float* b_mp3,
    const float* w2, const float* w3, const float* w4, const float* w5,
    const float* w6, const float* w7, const float* w8, const float* w9,
    const float* w10,
    __bf16* w2b, __bf16* w3b, __bf16* w4b, __bf16* w5b,
    __bf16* w6b, __bf16* w7b, __bf16* w8b, __bf16* w9b, __bf16* w10b,
    float* g, float* ab1, float* ab2, float* ab3)
{
  int o = blockIdx.x*256 + threadIdx.x;
  if (o < 4096)  { w2b[o] = (__bf16)w2[o]; return; } o -= 4096;
  if (o < 8192)  { w3b[o] = (__bf16)w3[o]; return; } o -= 8192;
  if (o < 4096)  { w4b[o] = (__bf16)w4[o]; return; } o -= 4096;
  if (o < 8192)  { w5b[o] = (__bf16)w5[o]; return; } o -= 8192;
  if (o < 196608){ w6b[o] = (__bf16)w6[o]; return; } o -= 196608;
  if (o < 311296){ w7b[o] = (__bf16)w7[o]; return; } o -= 311296;
  if (o < 65536) { w8b[o] = (__bf16)w8[o]; return; } o -= 65536;
  if (o < 32768) { w9b[o] = (__bf16)w9[o]; return; } o -= 32768;
  if (o < 8192)  { int r = o >> 7, c = o & 127;
                   w10b[o] = (__bf16)((r < 50) ? w10[r*128 + c] : 0.0f); return; } o -= 8192;
  if (o < 8192)  { g[o] = NINF; return; } o -= 8192;
  if (o < 3)  { float s = 0.f; for (int c=0;c<3;c++)  s += w_mp1[o*3+c];
                ab1[o] = s*s_mp1[o]; ab1[4+o]  = b_mp1[o]; return; } o -= 3;
  if (o < 64) { float s = 0.f; for (int c=0;c<64;c++) s += w_mp2[o*64+c];
                ab2[o] = s*s_mp2[o]; ab2[64+o] = b_mp2[o]; return; } o -= 64;
  if (o < 64) { float s = 0.f; for (int c=0;c<64;c++) s += w_mp3[o*64+c];
                ab3[o] = s*s_mp3[o]; ab3[64+o] = b_mp3[o]; return; }
}

// ---------------------------------------------------------------- knn on xyz (C=3)
__global__ __launch_bounds__(256) void knn_xyz_kernel(const float* in, int* idx)
{
  __shared__ float lp[NPTS*3];
  int b  = blockIdx.x >> 3;
  int rb = (blockIdx.x & 7) * 256;
  const float* xb = in + (long)b*6*NPTS;
  for (int i = threadIdx.x; i < NPTS*3; i += 256){
    int c = i >> 11, n = i & 2047;
    lp[n*3+c] = xb[c*NPTS + n];
  }
  __syncthreads();
  int row = rb + threadIdx.x;
  float xr = lp[row*3+0], yr = lp[row*3+1], zr = lp[row*3+2];
  float tv[KNN]; int ti[KNN];
#pragma unroll
  for (int k=0;k<KNN;k++){ tv[k]=NINF; ti[k]=0x7fffffff; }
  for (int j=0;j<NPTS;++j){
    float xj = lp[j*3], yj = lp[j*3+1], zj = lp[j*3+2];
    float sc = 2.f*(xr*xj + yr*yj + zr*zj) - (xj*xj + yj*yj + zj*zj);
    if (sc > tv[KNN-1]){
      tv[KNN-1]=sc; ti[KNN-1]=j;
#pragma unroll
      for (int p=KNN-1;p>0;--p){
        if (tv[p] > tv[p-1] || (tv[p] == tv[p-1] && ti[p] < ti[p-1])){
          float f=tv[p];tv[p]=tv[p-1];tv[p-1]=f;
          int z=ti[p];ti[p]=ti[p-1];ti[p-1]=z;
        }
      }
    }
  }
  int* op = idx + ((long)b*NPTS + row)*KNN;
#pragma unroll
  for (int k=0;k<KNN;k++) op[k]=ti[k];
}

// ---------------------------------------------------------------- manifold weight
__global__ __launch_bounds__(256) void weight_kernel(const float* in, const int* idx, float* wgt)
{
  int t = blockIdx.x*256 + threadIdx.x;
  int b = t >> 11, n = t & 2047;
  const float* nb = in + (long)b*6*NPTS + 3*NPTS;
  const int*   ip = idx + ((long)b*NPTS + n)*KNN;
  int j0 = ip[0];
  float cx = nb[j0], cy = nb[NPTS+j0], cz = nb[2*NPTS+j0];
  float cn = sqrtf(cx*cx + cy*cy + cz*cz);
  float* op = wgt + ((long)b*NPTS + n)*KNN;
  for (int k=0;k<KNN;++k){
    int j = ip[k];
    float gx = nb[j], gy = nb[NPTS+j], gz = nb[2*NPTS+j];
    float num = cx*gx + cy*gy + cz*gz;
    float den = fmaxf(cn*sqrtf(gx*gx+gy*gy+gz*gz), 1e-8f);
    op[k] = num/den;
  }
}

// ---------------------------------------------------------------- stage1: edge(xyz)+conv1(VALU)+conv2(WMMA)+maxK
__global__ __launch_bounds__(128) void stage1_kernel(
    const float* in, const int* idx, const float* wgt,
    const float* w1, const float* s1, const float* b1,
    const float* s2, const float* b2,
    const __bf16* w2b, const float* ab1,
    float* x1, __bf16* x1T, float* xxc)
{
  __shared__ float  lx[16][3];
  __shared__ int    lidx[16][KNN];
  __shared__ float  lw[16][KNN];
  __shared__ float  lw1[64*6];
  __shared__ float  ls1[64], lb1[64];
  __shared__ __bf16 h1b[64][72];
  __shared__ float  xmax[16][64];
  int tid = threadIdx.x;
  int b  = blockIdx.x >> 7;
  int nb = (blockIdx.x & 127) * 16;
  int wv = tid >> 5, lane = tid & 31, lh = lane >> 4;
  const float* xb = in + (long)b*6*NPTS;
  for (int i=tid;i<48;i+=128){ int p=i/3,c=i%3; lx[p][c]=xb[c*NPTS + nb + p]; }
  for (int i=tid;i<16*KNN;i+=128){ int p=i/KNN,k=i%KNN;
    lidx[p][k]=idx[((long)b*NPTS+nb+p)*KNN+k];
    lw[p][k]  =wgt[((long)b*NPTS+nb+p)*KNN+k]; }
  for (int i=tid;i<384;i+=128) lw1[i]=w1[i];
  if (tid<64){ ls1[tid]=s1[tid]; lb1[tid]=b1[tid]; }
  for (int i=tid;i<1024;i+=128) xmax[i>>6][i&63]=NINF;
  float a1[3], be1[3];
#pragma unroll
  for (int c=0;c<3;c++){ a1[c]=ab1[c]; be1[c]=ab1[4+c]; }
  // per-lane BN params for conv2 epilogue (oc = wv*16 + 8*lh + r)
  float s2r[8], b2r[8];
#pragma unroll
  for (int r=0;r<8;r++){ int oc = wv*16 + 8*lh + r; s2r[r]=s2[oc]; b2r[r]=b2[oc]; }
  __syncthreads();

  for (int chunk=0; chunk<5; ++chunk){
    { // build h1 columns (conv1, K=6, VALU)
      int col = tid & 63, grp = tid >> 6;
      int cg = chunk*64 + col;
      int p = cg/KNN, k = cg%KNN;
      int j = lidx[p][k];
      float w = lw[p][k];
      float e[6];
#pragma unroll
      for (int c=0;c<3;c++){
        float mod = lrelu(a1[c]*w + be1[c]);
        e[c]   = (xb[c*NPTS + j] - lx[p][c]) * mod;
        e[3+c] = lx[p][c];
      }
      for (int oc=grp*32; oc<grp*32+32; ++oc){
        float acc = 0.f;
#pragma unroll
        for (int c=0;c<6;c++) acc += lw1[oc*6+c]*e[c];
        h1b[col][oc] = (__bf16)lrelu(acc*ls1[oc] + lb1[oc]);
      }
    }
    __syncthreads();
    { // conv2 WMMA: M=64 (wave per 16-tile), K=64
      const __bf16* wbase = w2b + wv*16*64;
#pragma unroll
      for (int ct=0; ct<4; ++ct){
        v8f acc = {0.f,0.f,0.f,0.f,0.f,0.f,0.f,0.f};
#pragma unroll
        for (int ks=0; ks<64; ks+=32){
          v16bf A = load_tile16(wbase + ks, 64, lane);
          v16bf B = load_tile16(&h1b[ct*16][0] + ks, 72, lane);
          acc = wmma_bf16(A, B, acc);
        }
        int col = ct*16 + (lane & 15);
        int p = (chunk*64 + col)/KNN;
#pragma unroll
        for (int r=0;r<8;r++){
          int oc = wv*16 + 8*lh + r;
          dsMaxF(&xmax[p][oc], lrelu(acc[r]*s2r[r] + b2r[r]));
        }
      }
    }
    waitDS();
    __syncthreads();
  }
  for (int i=tid;i<1024;i+=128){
    int p=i>>6,c=i&63; float v=xmax[p][c];
    x1[(long)b*64*NPTS + (long)c*NPTS + nb + p] = v;
    x1T[((long)b*NPTS + nb + p)*64 + c] = (__bf16)v;
  }
  __syncthreads();
  if (tid<16){
    float s=0.f;
    for (int c=0;c<64;c++){ float v=xmax[tid][c]; s+=v*v; }
    xxc[(long)b*NPTS + nb + tid] = s;
  }
}

// ---------------------------------------------------------------- knn on 64-ch features via WMMA
__global__ __launch_bounds__(128) void knn_feat_kernel(const __bf16* xT, const float* xxc, int* idx)
{
  __shared__ float sc[16][256];
  __shared__ float pv[16][160];
  __shared__ int   pi[16][160];
  int b  = blockIdx.x >> 7;
  int rb = (blockIdx.x & 127) * 16;
  int tid = threadIdx.x, wv = tid>>5, lane = tid&31;
  const __bf16* xTb = xT + (long)b*NPTS*64;
  v16bf A0 = load_tile16(xTb + (long)rb*64 + 0, 64, lane);
  v16bf A1 = load_tile16(xTb + (long)rb*64 + 32, 64, lane);
  const float* xxb = xxc + (long)b*NPTS;
  int mrow = tid >> 3, msub = tid & 7;
  float tv[KNN]; int ti[KNN];
#pragma unroll
  for (int k=0;k<KNN;k++){ tv[k]=NINF; ti[k]=0x7fffffff; }
  for (int chunk=0; chunk<8; ++chunk){
    int cb = chunk*256;
#pragma unroll
    for (int i=0;i<4;i++){
      int ctile = wv*4 + i;
      int j0 = cb + ctile*16;
      v8f acc = {0.f,0.f,0.f,0.f,0.f,0.f,0.f,0.f};
      v16bf B0 = load_tile16(xTb + (long)j0*64 + 0, 64, lane);
      v16bf B1 = load_tile16(xTb + (long)j0*64 + 32, 64, lane);
      acc = wmma_bf16(A0, B0, acc);
      acc = wmma_bf16(A1, B1, acc);
      float xxv = xxb[j0 + (lane&15)];
#pragma unroll
      for (int r=0;r<8;r++){
        int row = r + 8*(lane>>4);
        sc[row][ctile*16 + (lane&15)] = 2.0f*acc[r] - xxv;
      }
    }
    __syncthreads();
    { // 8 threads per row, each merges a contiguous 32-col slice
      int q0 = msub*32;
      for (int q = q0; q < q0+32; ++q){
        float v = sc[mrow][q];
        int j = cb + q;
        if (v > tv[KNN-1]){
          tv[KNN-1]=v; ti[KNN-1]=j;
#pragma unroll
          for (int p=KNN-1;p>0;--p){
            if (tv[p] > tv[p-1] || (tv[p] == tv[p-1] && ti[p] < ti[p-1])){
              float f=tv[p];tv[p]=tv[p-1];tv[p-1]=f;
              int z=ti[p];ti[p]=ti[p-1];ti[p-1]=z;
            }
          }
        }
      }
    }
    __syncthreads();
  }
  // dump partial lists and do final 160-way merge per row
#pragma unroll
  for (int k=0;k<KNN;k++){ pv[mrow][msub*KNN+k]=tv[k]; pi[mrow][msub*KNN+k]=ti[k]; }
  __syncthreads();
  if (tid < 16){
    float fv[KNN]; int fi[KNN];
#pragma unroll
    for (int k=0;k<KNN;k++){ fv[k]=NINF; fi[k]=0x7fffffff; }
    for (int s=0;s<160;++s){
      float v = pv[tid][s]; int j = pi[tid][s];
      if (v > fv[KNN-1] || (v == fv[KNN-1] && j < fi[KNN-1])){
        fv[KNN-1]=v; fi[KNN-1]=j;
#pragma unroll
        for (int p=KNN-1;p>0;--p){
          if (fv[p] > fv[p-1] || (fv[p] == fv[p-1] && fi[p] < fi[p-1])){
            float f=fv[p];fv[p]=fv[p-1];fv[p-1]=f;
            int z=fi[p];fi[p]=fi[p-1];fi[p-1]=z;
          }
        }
      }
    }
    int* op = idx + ((long)b*NPTS + rb + tid)*KNN;
#pragma unroll
    for (int k=0;k<KNN;k++) op[k]=fi[k];
  }
}

// ---------------------------------------------------------------- stage2: edge(64)+conv3+conv4 (WMMA)+maxK
__global__ __launch_bounds__(128) void stage2_kernel(
    const float* xin, const int* idx, const float* wgt, const float* ab,
    const __bf16* wAb, const float* sA, const float* bA,
    const __bf16* wBb, const float* sB, const float* bB,
    float* xout, __bf16* xoutT, float* xxc)
{
  __shared__ float  lxf[16][64];
  __shared__ int    lidx[16][KNN];
  __shared__ float  lw[16][KNN];
  __shared__ __bf16 eB[64][136];
  __shared__ __bf16 h1b[64][72];
  __shared__ float  xmax[16][64];
  __shared__ float  la[64], lbt[64];
  int tid = threadIdx.x;
  int b  = blockIdx.x >> 7;
  int nb = (blockIdx.x & 127) * 16;
  int wv = tid>>5, lane = tid&31, lh = lane>>4;
  const float* xb = xin + (long)b*64*NPTS;
  for (int i=tid;i<1024;i+=128){ int p=i>>6,c=i&63;
    lxf[p][c]=xb[(long)c*NPTS + nb + p]; xmax[p][c]=NINF; }
  for (int i=tid;i<16*KNN;i+=128){ int p=i/KNN,k=i%KNN;
    lidx[p][k]=idx[((long)b*NPTS+nb+p)*KNN+k];
    lw[p][k]  =wgt[((long)b*NPTS+nb+p)*KNN+k]; }
  if (tid<64){ la[tid]=ab[tid]; lbt[tid]=ab[64+tid]; }
  // per-lane BN params (oc = wv*16 + 8*lh + r)
  float sAr[8], bAr[8], sBr[8], bBr[8];
#pragma unroll
  for (int r=0;r<8;r++){ int oc = wv*16 + 8*lh + r;
    sAr[r]=sA[oc]; bAr[r]=bA[oc]; sBr[r]=sB[oc]; bBr[r]=bB[oc]; }
  __syncthreads();
  for (int chunk=0; chunk<5; ++chunk){
    { // build edge columns (128 channels)
      int col = tid & 63, grp = tid >> 6;
      int cg = chunk*64 + col;
      int p = cg/KNN, k = cg%KNN;
      int j = lidx[p][k];
      float w = lw[p][k];
      if (grp == 0){
        for (int c=0;c<64;c++){
          float mod = lrelu(la[c]*w + lbt[c]);
          eB[col][c] = (__bf16)((xb[(long)c*NPTS + j] - lxf[p][c]) * mod);
        }
      } else {
        for (int c=0;c<64;c++) eB[col][64+c] = (__bf16)lxf[p][c];
      }
    }
    __syncthreads();
    { // conv3 WMMA: 64x128
      const __bf16* wbase = wAb + wv*16*128;
#pragma unroll
      for (int ct=0; ct<4; ++ct){
        v8f acc = {0.f,0.f,0.f,0.f,0.f,0.f,0.f,0.f};
#pragma unroll
        for (int ks=0; ks<128; ks+=32){
          v16bf A = load_tile16(wbase + ks, 128, lane);
          v16bf B = load_tile16(&eB[ct*16][0] + ks, 136, lane);
          acc = wmma_bf16(A, B, acc);
        }
        int col = ct*16 + (lane & 15);
#pragma unroll
        for (int r=0;r<8;r++){
          int oc = wv*16 + 8*lh + r;
          h1b[col][oc] = (__bf16)lrelu(acc[r]*sAr[r] + bAr[r]);
        }
      }
    }
    __syncthreads();
    { // conv4 WMMA: 64x64 + maxK
      const __bf16* wbase = wBb + wv*16*64;
#pragma unroll
      for (int ct=0; ct<4; ++ct){
        v8f acc = {0.f,0.f,0.f,0.f,0.f,0.f,0.f,0.f};
#pragma unroll
        for (int ks=0; ks<64; ks+=32){
          v16bf A = load_tile16(wbase + ks, 64, lane);
          v16bf B = load_tile16(&h1b[ct*16][0] + ks, 72, lane);
          acc = wmma_bf16(A, B, acc);
        }
        int col = ct*16 + (lane & 15);
        int p = (chunk*64 + col)/KNN;
#pragma unroll
        for (int r=0;r<8;r++){
          int oc = wv*16 + 8*lh + r;
          dsMaxF(&xmax[p][oc], lrelu(acc[r]*sBr[r] + bBr[r]));
        }
      }
    }
    waitDS();
    __syncthreads();
  }
  for (int i=tid;i<1024;i+=128){
    int p=i>>6,c=i&63; float v=xmax[p][c];
    xout[(long)b*64*NPTS + (long)c*NPTS + nb + p] = v;
    xoutT[((long)b*NPTS + nb + p)*64 + c] = (__bf16)v;
  }
  __syncthreads();
  if (tid<16){
    float s=0.f;
    for (int c=0;c<64;c++){ float v=xmax[tid][c]; s+=v*v; }
    xxc[(long)b*NPTS + nb + tid] = s;
  }
}

// ---------------------------------------------------------------- stage3: edge(64)+conv5 (WMMA)+maxK
__global__ __launch_bounds__(128) void stage3_kernel(
    const float* xin, const int* idx, const float* wgt, const float* ab,
    const __bf16* wAb, const float* sA, const float* bA,
    float* xout, __bf16* xoutT)
{
  __shared__ float  lxf[16][64];
  __shared__ int    lidx[16][KNN];
  __shared__ float  lw[16][KNN];
  __shared__ __bf16 eB[64][136];
  __shared__ float  xmax[16][64];
  __shared__ float  la[64], lbt[64];
  int tid = threadIdx.x;
  int b  = blockIdx.x >> 7;
  int nb = (blockIdx.x & 127) * 16;
  int wv = tid>>5, lane = tid&31, lh = lane>>4;
  const float* xb = xin + (long)b*64*NPTS;
  for (int i=tid;i<1024;i+=128){ int p=i>>6,c=i&63;
    lxf[p][c]=xb[(long)c*NPTS + nb + p]; xmax[p][c]=NINF; }
  for (int i=tid;i<16*KNN;i+=128){ int p=i/KNN,k=i%KNN;
    lidx[p][k]=idx[((long)b*NPTS+nb+p)*KNN+k];
    lw[p][k]  =wgt[((long)b*NPTS+nb+p)*KNN+k]; }
  if (tid<64){ la[tid]=ab[tid]; lbt[tid]=ab[64+tid]; }
  float sAr[8], bAr[8];
#pragma unroll
  for (int r=0;r<8;r++){ int oc = wv*16 + 8*lh + r; sAr[r]=sA[oc]; bAr[r]=bA[oc]; }
  __syncthreads();
  for (int chunk=0; chunk<5; ++chunk){
    {
      int col = tid & 63, grp = tid >> 6;
      int cg = chunk*64 + col;
      int p = cg/KNN, k = cg%KNN;
      int j = lidx[p][k];
      float w = lw[p][k];
      if (grp == 0){
        for (int c=0;c<64;c++){
          float mod = lrelu(la[c]*w + lbt[c]);
          eB[col][c] = (__bf16)((xb[(long)c*NPTS + j] - lxf[p][c]) * mod);
        }
      } else {
        for (int c=0;c<64;c++) eB[col][64+c] = (__bf16)lxf[p][c];
      }
    }
    __syncthreads();
    {
      const __bf16* wbase = wAb + wv*16*128;
#pragma unroll
      for (int ct=0; ct<4; ++ct){
        v8f acc = {0.f,0.f,0.f,0.f,0.f,0.f,0.f,0.f};
#pragma unroll
        for (int ks=0; ks<128; ks+=32){
          v16bf A = load_tile16(wbase + ks, 128, lane);
          v16bf B = load_tile16(&eB[ct*16][0] + ks, 136, lane);
          acc = wmma_bf16(A, B, acc);
        }
        int col = ct*16 + (lane & 15);
        int p = (chunk*64 + col)/KNN;
#pragma unroll
        for (int r=0;r<8;r++){
          int oc = wv*16 + 8*lh + r;
          dsMaxF(&xmax[p][oc], lrelu(acc[r]*sAr[r] + bAr[r]));
        }
      }
    }
    waitDS();
    __syncthreads();
  }
  for (int i=tid;i<1024;i+=128){
    int p=i>>6,c=i&63; float v=xmax[p][c];
    xout[(long)b*64*NPTS + (long)c*NPTS + nb + p] = v;
    xoutT[((long)b*NPTS + nb + p)*64 + c] = (__bf16)v;
  }
}

// ---------------------------------------------------------------- conv6 (1024x192 WMMA) + global max
__global__ __launch_bounds__(256) void conv6_kernel(
    const __bf16* x1T, const __bf16* x2T, const __bf16* x3T,
    const __bf16* w6b, const float* s6, const float* b6, float* g)
{
  __shared__ __bf16 B6[64][200];
  __shared__ float  gloc[1024];
  int tid = threadIdx.x;
  int b  = blockIdx.x >> 5;
  int nb = (blockIdx.x & 31) * 64;
  for (int i=tid;i<1024;i+=256) gloc[i]=NINF;
  // async global->LDS staging: 24 x 16B segments per row
  for (int i=tid;i<64*24;i+=256){
    int p = i/24, seg = i%24;
    int cc = seg*8;
    long base = ((long)b*NPTS + nb + p)*64;
    const __bf16* src;
    if (cc < 64)       src = x1T + base + cc;
    else if (cc < 128) src = x2T + base + (cc-64);
    else               src = x3T + base + (cc-128);
    unsigned loff = (unsigned)(size_t)(&B6[0][0]) + (unsigned)(p*200 + cc)*2u;
    async_g2l_b128(loff, src);
  }
  waitAsync();
  __syncthreads();
  int wv = tid>>5, lane = tid&31, lh = lane>>4;
  for (int mi=0; mi<8; ++mi){
    int mt = wv + mi*8;
    const __bf16* wbase = w6b + (long)mt*16*192;
    // hoist per-lane BN params for this M-tile out of the col-tile loop
    float s6r[8], b6r[8];
#pragma unroll
    for (int r=0;r<8;r++){ int oc = mt*16 + 8*lh + r; s6r[r]=s6[oc]; b6r[r]=b6[oc]; }
    for (int ct=0; ct<4; ++ct){
      v8f acc = {0.f,0.f,0.f,0.f,0.f,0.f,0.f,0.f};
#pragma unroll
      for (int ks=0; ks<192; ks+=32){
        v16bf A = load_tile16(wbase + ks, 192, lane);
        v16bf B = load_tile16(&B6[ct*16][0] + ks, 200, lane);
        acc = wmma_bf16(A, B, acc);
      }
#pragma unroll
      for (int r=0;r<8;r++){
        int oc = mt*16 + 8*lh + r;
        dsMaxF(&gloc[oc], lrelu(acc[r]*s6r[r] + b6r[r]));
      }
    }
  }
  waitDS();
  __syncthreads();
  for (int i=tid;i<1024;i+=256) globalMaxF(&g[(long)b*1024 + i], gloc[i]);
}

__global__ __launch_bounds__(256) void g2bf_kernel(const float* g, __bf16* gbf){
  int t = blockIdx.x*256 + threadIdx.x;
  if (t < BATCH*1024) gbf[t] = (__bf16)g[t];
}

// ---------------------------------------------------------------- head: conv7->conv8->conv9->w10, fully fused
__global__ __launch_bounds__(256) void head_kernel(
    const __bf16* gbf, const __bf16* x1T, const __bf16* x2T, const __bf16* x3T,
    const __bf16* w7b, const float* s7, const float* b7,
    const __bf16* w8b, const float* s8, const float* b8,
    const __bf16* w9b, const float* s9, const float* b9,
    const __bf16* w10b, float* out)
{
  __shared__ __bf16 B7[16][1224];
  __shared__ __bf16 h7[16][264];
  __shared__ __bf16 h8[16][264];
  __shared__ __bf16 h9[16][136];
  int tid = threadIdx.x;
  int b  = blockIdx.x >> 7;
  int nb = (blockIdx.x & 127) * 16;
  // async global->LDS staging: 152 x 16B segments per row (128 g + 8+8+8 x)
  for (int i=tid;i<16*152;i+=256){
    int p = i/152, seg = i%152;
    int cc = seg*8;
    const __bf16* src;
    if (cc < 1024) src = gbf + (long)b*1024 + cc;
    else {
      long base = ((long)b*NPTS + nb + p)*64;
      int c2 = cc - 1024;
      if (c2 < 64)       src = x1T + base + c2;
      else if (c2 < 128) src = x2T + base + (c2-64);
      else               src = x3T + base + (c2-128);
    }
    unsigned loff = (unsigned)(size_t)(&B7[0][0]) + (unsigned)(p*1224 + cc)*2u;
    async_g2l_b128(loff, src);
  }
  waitAsync();
  __syncthreads();
  int wv = tid>>5, lane = tid&31;
  int col = lane & 15, lh = lane >> 4;
  // conv7: M=256 (2 tiles/wave), K=1216
  for (int mi=0; mi<2; ++mi){
    int mt = wv*2 + mi;
    const __bf16* wbase = w7b + (long)mt*16*1216;
    v8f acc = {0.f,0.f,0.f,0.f,0.f,0.f,0.f,0.f};
    for (int ks=0; ks<1216; ks+=32){
      v16bf A = load_tile16(wbase + ks, 1216, lane);
      v16bf B = load_tile16(&B7[0][0] + ks, 1224, lane);
      acc = wmma_bf16(A, B, acc);
    }
#pragma unroll
    for (int r=0;r<8;r++){
      int oc = mt*16 + 8*lh + r;
      h7[col][oc] = (__bf16)lrelu(acc[r]*s7[oc] + b7[oc]);
    }
  }
  __syncthreads();
  // conv8: M=256, K=256
  for (int mi=0; mi<2; ++mi){
    int mt = wv*2 + mi;
    const __bf16* wbase = w8b + (long)mt*16*256;
    v8f acc = {0.f,0.f,0.f,0.f,0.f,0.f,0.f,0.f};
#pragma unroll
    for (int ks=0; ks<256; ks+=32){
      v16bf A = load_tile16(wbase + ks, 256, lane);
      v16bf B = load_tile16(&h7[0][0] + ks, 264, lane);
      acc = wmma_bf16(A, B, acc);
    }
#pragma unroll
    for (int r=0;r<8;r++){
      int oc = mt*16 + 8*lh + r;
      h8[col][oc] = (__bf16)lrelu(acc[r]*s8[oc] + b8[oc]);
    }
  }
  __syncthreads();
  // conv9: M=128 (1 tile/wave), K=256
  {
    int mt = wv;
    const __bf16* wbase = w9b + (long)mt*16*256;
    v8f acc = {0.f,0.f,0.f,0.f,0.f,0.f,0.f,0.f};
#pragma unroll
    for (int ks=0; ks<256; ks+=32){
      v16bf A = load_tile16(wbase + ks, 256, lane);
      v16bf B = load_tile16(&h8[0][0] + ks, 264, lane);
      acc = wmma_bf16(A, B, acc);
    }
#pragma unroll
    for (int r=0;r<8;r++){
      int oc = mt*16 + 8*lh + r;
      h9[col][oc] = (__bf16)lrelu(acc[r]*s9[oc] + b9[oc]);
    }
  }
  __syncthreads();
  // conv10: M=64 (padded from 50), K=128, no activation
  if (wv < 4){
    int mt = wv;
    const __bf16* wbase = w10b + (long)mt*16*128;
    v8f acc = {0.f,0.f,0.f,0.f,0.f,0.f,0.f,0.f};
#pragma unroll
    for (int ks=0; ks<128; ks+=32){
      v16bf A = load_tile16(wbase + ks, 128, lane);
      v16bf B = load_tile16(&h9[0][0] + ks, 136, lane);
      acc = wmma_bf16(A, B, acc);
    }
#pragma unroll
    for (int r=0;r<8;r++){
      int o = mt*16 + 8*lh + r;
      if (o < 50)
        out[(long)b*50*NPTS + (long)o*NPTS + nb + col] = acc[r];
    }
  }
}

// ---------------------------------------------------------------- launch
extern "C" void kernel_launch(void* const* d_in, const int* in_sizes, int n_in,
                              void* d_out, int out_size, void* d_ws, size_t ws_size,
                              hipStream_t stream)
{
  const float* xin   = (const float*)d_in[0];
  const float* w_mp1 = (const float*)d_in[1];
  const float* s_mp1 = (const float*)d_in[2];
  const float* b_mp1 = (const float*)d_in[3];
  const float* w_mp2 = (const float*)d_in[4];
  const float* s_mp2 = (const float*)d_in[5];
  const float* b_mp2 = (const float*)d_in[6];
  const float* w_mp3 = (const float*)d_in[7];
  const float* s_mp3 = (const float*)d_in[8];
  const float* b_mp3 = (const float*)d_in[9];
  const float* w1 = (const float*)d_in[10]; const float* s1 = (const float*)d_in[11]; const float* b1 = (const float*)d_in[12];
  const float* w2 = (const float*)d_in[13]; const float* s2 = (const float*)d_in[14]; const float* b2 = (const float*)d_in[15];
  const float* w3 = (const float*)d_in[16]; const float* s3 = (const float*)d_in[17]; const float* b3 = (const float*)d_in[18];
  const float* w4 = (const float*)d_in[19]; const float* s4 = (const float*)d_in[20]; const float* b4 = (const float*)d_in[21];
  const float* w5 = (const float*)d_in[22]; const float* s5 = (const float*)d_in[23]; const float* b5 = (const float*)d_in[24];
  const float* w6 = (const float*)d_in[25]; const float* s6 = (const float*)d_in[26]; const float* b6 = (const float*)d_in[27];
  const float* w7 = (const float*)d_in[28]; const float* s7 = (const float*)d_in[29]; const float* b7 = (const float*)d_in[30];
  const float* w8 = (const float*)d_in[31]; const float* s8 = (const float*)d_in[32]; const float* b8 = (const float*)d_in[33];
  const float* w9 = (const float*)d_in[34]; const float* s9 = (const float*)d_in[35]; const float* b9 = (const float*)d_in[36];
  const float* w10 = (const float*)d_in[37];
  float* out = (float*)d_out;

  char* ws = (char*)d_ws;
  size_t off = 0;
  auto take = [&](size_t n) -> void* {
    void* p = ws + off;
    off += (n + 255) & ~(size_t)255;
    return p;
  };
  int*    idx  = (int*)   take((size_t)BATCH*NPTS*KNN*4);
  float*  wgt  = (float*) take((size_t)BATCH*NPTS*KNN*4);
  float*  x1   = (float*) take((size_t)BATCH*64*NPTS*4);
  float*  x2   = (float*) take((size_t)BATCH*64*NPTS*4);
  float*  x3   = (float*) take((size_t)BATCH*64*NPTS*4);
  __bf16* x1T  = (__bf16*)take((size_t)BATCH*NPTS*64*2);
  __bf16* x2T  = (__bf16*)take((size_t)BATCH*NPTS*64*2);
  __bf16* x3T  = (__bf16*)take((size_t)BATCH*NPTS*64*2);
  float*  xxc  = (float*) take((size_t)BATCH*NPTS*4);
  float*  g    = (float*) take((size_t)BATCH*1024*4);
  __bf16* gbf  = (__bf16*)take((size_t)BATCH*1024*2);
  __bf16* w2b  = (__bf16*)take(4096*2);
  __bf16* w3b  = (__bf16*)take(8192*2);
  __bf16* w4b  = (__bf16*)take(4096*2);
  __bf16* w5b  = (__bf16*)take(8192*2);
  __bf16* w6b  = (__bf16*)take(196608*2);
  __bf16* w7b  = (__bf16*)take(311296*2);
  __bf16* w8b  = (__bf16*)take(65536*2);
  __bf16* w9b  = (__bf16*)take(32768*2);
  __bf16* w10b = (__bf16*)take(8192*2);
  float*  ab1  = (float*) take(8*4);
  float*  ab2  = (float*) take(128*4);
  float*  ab3  = (float*) take(128*4);
  (void)in_sizes; (void)n_in; (void)out_size; (void)ws_size;

  prep_kernel<<<2529, 256, 0, stream>>>(w_mp1,s_mp1,b_mp1,w_mp2,s_mp2,b_mp2,w_mp3,s_mp3,b_mp3,
                                        w2,w3,w4,w5,w6,w7,w8,w9,w10,
                                        w2b,w3b,w4b,w5b,w6b,w7b,w8b,w9b,w10b,
                                        g,ab1,ab2,ab3);
  knn_xyz_kernel<<<BATCH*(NPTS/256), 256, 0, stream>>>(xin, idx);
  weight_kernel<<<(BATCH*NPTS)/256, 256, 0, stream>>>(xin, idx, wgt);
  stage1_kernel<<<BATCH*(NPTS/16), 128, 0, stream>>>(xin, idx, wgt, w1, s1, b1, s2, b2,
                                                     w2b, ab1, x1, x1T, xxc);
  knn_feat_kernel<<<BATCH*(NPTS/16), 128, 0, stream>>>(x1T, xxc, idx);
  stage2_kernel<<<BATCH*(NPTS/16), 128, 0, stream>>>(x1, idx, wgt, ab2,
                                                     w3b, s3, b3, w4b, s4, b4,
                                                     x2, x2T, xxc);
  knn_feat_kernel<<<BATCH*(NPTS/16), 128, 0, stream>>>(x2T, xxc, idx);
  stage3_kernel<<<BATCH*(NPTS/16), 128, 0, stream>>>(x2, idx, wgt, ab3,
                                                     w5b, s5, b5, x3, x3T);
  conv6_kernel<<<BATCH*(NPTS/64), 256, 0, stream>>>(x1T, x2T, x3T, w6b, s6, b6, g);
  g2bf_kernel<<<(BATCH*1024)/256, 256, 0, stream>>>(g, gbf);
  head_kernel<<<BATCH*(NPTS/16), 256, 0, stream>>>(gbf, x1T, x2T, x3T,
                                                   w7b, s7, b7, w8b, s8, b8,
                                                   w9b, s9, b9, w10b, out);
}
